// SpikeDrivenSelfAttention_33019708571663
// MI455X (gfx1250) — compile-verified, matrix operand order checked by reading, not run on hardware
//
#include <hip/hip_runtime.h>
#include <hip/hip_bf16.h>
#include <cstdint>

typedef _Float16 v8h  __attribute__((ext_vector_type(8)));
typedef _Float16 v16h __attribute__((ext_vector_type(16)));
typedef float    v8f  __attribute__((ext_vector_type(8)));

#define T_DIM 4
#define B_DIM 32
#define C_DIM 256
#define N_PIX 1024
#define BNC   (B_DIM * N_PIX * C_DIM)   // 8388608 elements per (matrix, t-slice set)

// ---- workspace layout (bytes) ----
#define XS_OFF   0ull                                        // xs_t  f16 [T][B][N][C]      64 MB
#define U_OFF    (XS_OFF  + 2ull * T_DIM * BNC)              // u_qkv f16 3x[T][B][N][C]   192 MB
#define WH_OFF   (U_OFF   + 3ull * 2ull * T_DIM * BNC)       // wq,wk,wv,wp f16            512 KB
#define KVP_OFF  (WH_OFF  + 4ull * 2ull * C_DIM * C_DIM)     // kv partials f32              1 MB
#define KVS_OFF  (KVP_OFF + 4ull * T_DIM * B_DIM * 8 * C_DIM)// kv spikes f16               64 KB

// ---------------- fragment loaders (ISA §7.12.2 layouts) ----------------
// A (16x32 f16): lane l<16 holds row M=l, K {0..7} in halves 0..7 and K {16..23}
// in halves 8..15; lanes 16..31 hold K {8..15} / {24..31}.  B (32x16) mirrored
// with lane = column; since B[k][n] = w[n][k] and w is row-major [o][c], both
// operands load as two contiguous 16-byte chunks from a row.
__device__ inline v16h load_frag16(const _Float16* __restrict__ row, int kBase, int hs) {
    const v8h lo = *(const v8h*)(row + kBase + 8 * hs);
    const v8h hi = *(const v8h*)(row + kBase + 16 + 8 * hs);
    v16h r;
#pragma unroll
    for (int j = 0; j < 8; ++j) { r[j] = lo[j]; r[j + 8] = hi[j]; }
    return r;
}

__device__ inline v16h load_frag16_scaled(const _Float16* __restrict__ row, int kBase, int hs,
                                          const _Float16* kvsh) {
    const v8h lo = *(const v8h*)(row + kBase + 8 * hs);
    const v8h hi = *(const v8h*)(row + kBase + 16 + 8 * hs);
    v16h r;
#pragma unroll
    for (int j = 0; j < 8; ++j) {
        r[j]     = (_Float16)(lo[j] * kvsh[kBase + 8 * hs + j]);
        r[j + 8] = (_Float16)(hi[j] * kvsh[kBase + 16 + 8 * hs + j]);
    }
    return r;
}

// ---------------- K0: convert weights f32 -> f16 ----------------
__global__ void sdsa_wcvt(const float* __restrict__ wq, const float* __restrict__ wk,
                          const float* __restrict__ wv, const float* __restrict__ wp,
                          _Float16* __restrict__ wh) {
    int j = blockIdx.x * 256 + threadIdx.x;                  // < 65536
    const float* src = (blockIdx.y == 0) ? wq : (blockIdx.y == 1) ? wk
                      : (blockIdx.y == 2) ? wv : wp;
    wh[(size_t)blockIdx.y * (C_DIM * C_DIM) + j] = (_Float16)src[j];
}

// ---------------- K1: shortcut LIF + transpose to [t,b,n,c] f16 ----------------
__global__ void sdsa_lif_in(const float* __restrict__ x, _Float16* __restrict__ xs) {
    const int tx = threadIdx.x, ty = threadIdx.y;            // block 32x8
    const int nb = blockIdx.x * 32, cb = blockIdx.y * 32, b = blockIdx.z;
    __shared__ _Float16 tile[32 * 33];                       // [c][n] padded
    float v[4] = {0.f, 0.f, 0.f, 0.f};
    for (int t = 0; t < T_DIM; ++t) {
#pragma unroll
        for (int i = 0; i < 4; ++i) {
            int c = cb + ty + 8 * i;
            float xv = x[((size_t)(t * B_DIM + b) * C_DIM + c) * N_PIX + nb + tx];
            float h = 0.5f * (v[i] + xv);                    // tau = 2
            float s = (h >= 1.0f) ? 1.0f : 0.0f;
            v[i] = (s != 0.f) ? 0.f : h;                     // hard reset
            tile[(ty + 8 * i) * 33 + tx] = (_Float16)s;
        }
        __syncthreads();
        int flat = ty * 32 + tx, cc = flat & 31, rr = flat >> 5;
#pragma unroll
        for (int p = 0; p < 4; ++p) {
            int nn = rr + 8 * p;
            xs[((size_t)(t * B_DIM + b) * N_PIX + nb + nn) * C_DIM + cb + cc] = tile[cc * 33 + nn];
        }
        __syncthreads();
    }
}

// ---------------- K2: q/k/v GEMM + folded BN -> f16 pre-activations ----------------
// block = 8 waves (4 M x 2 N), block tile 128x64, wave tile 32x32 (2x2 WMMA tiles)
__global__ __launch_bounds__(256) void sdsa_gemm_qkv(
    const _Float16* __restrict__ xs, const _Float16* __restrict__ wh, _Float16* __restrict__ u,
    const float* __restrict__ qs, const float* __restrict__ qb,
    const float* __restrict__ ks, const float* __restrict__ kb2,
    const float* __restrict__ vs, const float* __restrict__ vb) {
    const int z = blockIdx.z;
    const int mat = z / (T_DIM * B_DIM);
    const int rem = z % (T_DIM * B_DIM);
    const size_t tb = (size_t)rem;                           // t*B + b
    const _Float16* wmat = wh + (size_t)mat * C_DIM * C_DIM;
    _Float16* umat = u + (size_t)mat * T_DIM * BNC;
    const float* sc = (mat == 0) ? qs : (mat == 1) ? ks : vs;
    const float* bi = (mat == 0) ? qb : (mat == 1) ? kb2 : vb;

    const int lane = threadIdx.x & 31, wave = threadIdx.x >> 5;
    const int wm = wave & 3, wn = wave >> 2;
    const int hs = lane >> 4, l15 = lane & 15;
    const int m0 = blockIdx.x * 128 + wm * 32;
    const int n0 = blockIdx.y * 64 + wn * 32;

    const _Float16* Arow[2];
    const _Float16* Brow[2];
#pragma unroll
    for (int s = 0; s < 2; ++s) {
        Arow[s] = xs + (tb * N_PIX + m0 + s * 16 + l15) * C_DIM;
        Brow[s] = wmat + (size_t)(n0 + s * 16 + l15) * C_DIM;
    }

    v8f acc[2][2] = {};
#pragma unroll
    for (int kb = 0; kb < 8; ++kb) {
        const int kBase = kb * 32;
        v16h a0 = load_frag16(Arow[0], kBase, hs);
        v16h a1 = load_frag16(Arow[1], kBase, hs);
        v16h b0 = load_frag16(Brow[0], kBase, hs);
        v16h b1 = load_frag16(Brow[1], kBase, hs);
        acc[0][0] = __builtin_amdgcn_wmma_f32_16x16x32_f16(false, a0, false, b0, (short)0, acc[0][0], false, false);
        acc[0][1] = __builtin_amdgcn_wmma_f32_16x16x32_f16(false, a0, false, b1, (short)0, acc[0][1], false, false);
        acc[1][0] = __builtin_amdgcn_wmma_f32_16x16x32_f16(false, a1, false, b0, (short)0, acc[1][0], false, false);
        acc[1][1] = __builtin_amdgcn_wmma_f32_16x16x32_f16(false, a1, false, b1, (short)0, acc[1][1], false, false);
    }
#pragma unroll
    for (int ms = 0; ms < 2; ++ms)
#pragma unroll
        for (int ns = 0; ns < 2; ++ns) {
            int o = n0 + ns * 16 + l15;
            float scv = sc[o], biv = bi[o];
#pragma unroll
            for (int r = 0; r < 8; ++r) {
                int m = m0 + ms * 16 + r + 8 * hs;
                umat[(tb * N_PIX + m) * C_DIM + o] = (_Float16)(acc[ms][ns][r] * scv + biv);
            }
        }
}

// ---------------- K2b: LIF scan over T on q/k/v pre-activations (in place) ----------------
__global__ void sdsa_lif_qkv(_Float16* __restrict__ u) {
    size_t g = (size_t)blockIdx.x * 256 + threadIdx.x;       // < 3*BNC
    size_t mat = g / BNC, e = g % BNC;
    _Float16* base = u + mat * (size_t)T_DIM * BNC + e;
    float v = 0.f;
#pragma unroll
    for (int t = 0; t < T_DIM; ++t) {
        float uv = (float)base[(size_t)t * BNC];
        float h = 0.5f * (v + uv);
        float s = (h >= 1.0f) ? 1.0f : 0.0f;
        base[(size_t)t * BNC] = (_Float16)s;
        v = (s != 0.f) ? 0.f : h;
    }
}

// ---------------- K3a: kv partial reduction over N ----------------
__global__ void sdsa_kv_reduce(const _Float16* __restrict__ u, float* __restrict__ part) {
    const int tb = blockIdx.x;                               // t*B + b
    const int sl = blockIdx.y;                               // 0..7, 128 pixels each
    const int o = threadIdx.x;                               // channel
    const _Float16* kp = u + 1ull * T_DIM * BNC;
    const _Float16* vp = u + 2ull * T_DIM * BNC;
    float s = 0.f;
    for (int j = 0; j < 128; ++j) {
        size_t idx = ((size_t)tb * N_PIX + sl * 128 + j) * C_DIM + o;
        s += (float)kp[idx] * (float)vp[idx];
    }
    part[((size_t)tb * 8 + sl) * C_DIM + o] = s;
}

// ---------------- K3b: kv LIF (v_th = 0.5) ----------------
__global__ void sdsa_kv_lif(const float* __restrict__ part, _Float16* __restrict__ kvs) {
    int g = blockIdx.x * 256 + threadIdx.x;                  // < B*C
    int b = g / C_DIM, o = g % C_DIM;
    float v = 0.f;
#pragma unroll
    for (int t = 0; t < T_DIM; ++t) {
        int tb = t * B_DIM + b;
        float s8 = 0.f;
#pragma unroll
        for (int sl = 0; sl < 8; ++sl) s8 += part[((size_t)tb * 8 + sl) * C_DIM + o];
        float h = 0.5f * (v + s8);
        float sp = (h >= 0.5f) ? 1.0f : 0.0f;
        kvs[(size_t)tb * C_DIM + o] = (_Float16)sp;
        v = (sp != 0.f) ? 0.f : h;
    }
}

// ---------------- K4: projection GEMM with A = q*kv, epilogue BN+bias+identity ----------------
__global__ __launch_bounds__(256) void sdsa_gemm_proj(
    const _Float16* __restrict__ qsp, const _Float16* __restrict__ kvsp,
    const _Float16* __restrict__ wph, const float* __restrict__ x,
    const float* __restrict__ bp, const float* __restrict__ pscale,
    const float* __restrict__ pbias, float* __restrict__ out) {
    const size_t tb = (size_t)blockIdx.z;                    // t*B + b
    __shared__ _Float16 kvsh[C_DIM];
    __shared__ float lt[64 * 129];                           // [o2_local][n_local], padded
    kvsh[threadIdx.x] = kvsp[tb * C_DIM + threadIdx.x];
    __syncthreads();

    const int lane = threadIdx.x & 31, wave = threadIdx.x >> 5;
    const int wm = wave & 3, wn = wave >> 2;
    const int hs = lane >> 4, l15 = lane & 15;
    const int m0 = blockIdx.x * 128 + wm * 32;               // pixel rows
    const int n0 = blockIdx.y * 64 + wn * 32;                // output channels

    const _Float16* Arow[2];
    const _Float16* Brow[2];
#pragma unroll
    for (int s = 0; s < 2; ++s) {
        Arow[s] = qsp + (tb * N_PIX + m0 + s * 16 + l15) * C_DIM;
        Brow[s] = wph + (size_t)(n0 + s * 16 + l15) * C_DIM;
    }

    v8f acc[2][2] = {};
#pragma unroll
    for (int kb = 0; kb < 8; ++kb) {
        const int kBase = kb * 32;
        v16h a0 = load_frag16_scaled(Arow[0], kBase, hs, kvsh);
        v16h a1 = load_frag16_scaled(Arow[1], kBase, hs, kvsh);
        v16h b0 = load_frag16(Brow[0], kBase, hs);
        v16h b1 = load_frag16(Brow[1], kBase, hs);
        acc[0][0] = __builtin_amdgcn_wmma_f32_16x16x32_f16(false, a0, false, b0, (short)0, acc[0][0], false, false);
        acc[0][1] = __builtin_amdgcn_wmma_f32_16x16x32_f16(false, a0, false, b1, (short)0, acc[0][1], false, false);
        acc[1][0] = __builtin_amdgcn_wmma_f32_16x16x32_f16(false, a1, false, b0, (short)0, acc[1][0], false, false);
        acc[1][1] = __builtin_amdgcn_wmma_f32_16x16x32_f16(false, a1, false, b1, (short)0, acc[1][1], false, false);
    }

    // stash tile transposed in LDS: row = output channel (local), col = pixel (local)
#pragma unroll
    for (int ms = 0; ms < 2; ++ms)
#pragma unroll
        for (int ns = 0; ns < 2; ++ns) {
            int ol = wn * 32 + ns * 16 + l15;
#pragma unroll
            for (int r = 0; r < 8; ++r) {
                int ml = wm * 32 + ms * 16 + r + 8 * hs;
                lt[ol * 129 + ml] = acc[ms][ns][r];
            }
        }
    __syncthreads();

    // coalesced stores in reference [t,b,c,n] layout, fused bias/BN/identity
    for (int idx = threadIdx.x; idx < 64 * 128; idx += 256) {
        int rr = idx >> 7, cn = idx & 127;
        int o2 = blockIdx.y * 64 + rr;
        int n  = blockIdx.x * 128 + cn;
        float scv = pscale[o2];
        float val = lt[rr * 129 + cn] * scv + (bp[o2] * scv + pbias[o2]);
        size_t oi = (tb * C_DIM + o2) * N_PIX + n;
        out[oi] = val + x[oi];
    }
}

extern "C" void kernel_launch(void* const* d_in, const int* in_sizes, int n_in,
                              void* d_out, int out_size, void* d_ws, size_t ws_size,
                              hipStream_t stream) {
    (void)in_sizes; (void)n_in; (void)out_size; (void)ws_size;
    const float* x       = (const float*)d_in[0];
    const float* wq      = (const float*)d_in[1];
    const float* wk      = (const float*)d_in[2];
    const float* wv      = (const float*)d_in[3];
    const float* wp      = (const float*)d_in[4];
    const float* bp      = (const float*)d_in[5];
    const float* q_scale = (const float*)d_in[6];
    const float* q_bias  = (const float*)d_in[7];
    const float* k_scale = (const float*)d_in[8];
    const float* k_bias  = (const float*)d_in[9];
    const float* v_scale = (const float*)d_in[10];
    const float* v_bias  = (const float*)d_in[11];
    const float* p_scale = (const float*)d_in[12];
    const float* p_bias  = (const float*)d_in[13];

    char* ws = (char*)d_ws;
    _Float16* xs   = (_Float16*)(ws + XS_OFF);
    _Float16* u    = (_Float16*)(ws + U_OFF);
    _Float16* wh   = (_Float16*)(ws + WH_OFF);
    float*    part = (float*)(ws + KVP_OFF);
    _Float16* kvs  = (_Float16*)(ws + KVS_OFF);
    float*    out  = (float*)d_out;

    sdsa_wcvt<<<dim3(256, 4), 256, 0, stream>>>(wq, wk, wv, wp, wh);
    sdsa_lif_in<<<dim3(32, 8, 32), dim3(32, 8), 0, stream>>>(x, xs);
    sdsa_gemm_qkv<<<dim3(8, 4, 3 * T_DIM * B_DIM), 256, 0, stream>>>(
        xs, wh, u, q_scale, q_bias, k_scale, k_bias, v_scale, v_bias);
    sdsa_lif_qkv<<<dim3(3u * BNC / 256u), 256, 0, stream>>>(u);
    sdsa_kv_reduce<<<dim3(T_DIM * B_DIM, 8), 256, 0, stream>>>(u, part);
    sdsa_kv_lif<<<dim3(B_DIM * C_DIM / 256), 256, 0, stream>>>(part, kvs);
    sdsa_gemm_proj<<<dim3(8, 4, T_DIM * B_DIM), 256, 0, stream>>>(
        u /* q spikes (mat 0) */, kvs, wh + 3 * C_DIM * C_DIM, x, bp, p_scale, p_bias, out);
}
